// PointTransformer_42563125903631
// MI455X (gfx1250) — matrix operand order repaired and checked by reading, not compiled, not run on hardware
//
#include <hip/hip_runtime.h>
#include <hip/hip_bf16.h>
#include <stdint.h>
#include <stddef.h>

// ============================================================================
// Types for CDNA5 WMMA
// ============================================================================
typedef __attribute__((ext_vector_type(16))) __bf16 v16bf;
typedef __attribute__((ext_vector_type(8)))  __bf16 v8bf;
typedef __attribute__((ext_vector_type(2)))  __bf16 v2bf;
typedef __attribute__((ext_vector_type(8)))  float  v8f;

#define BN_EPS 1e-5f

// ============================================================================
// GEMM: C[M,N] = A[M,K] @ W[K,N] (+bias) (optional relu), via bf16 WMMA.
//
// Block = 256 threads = 8 waves; macro-tile BM(128) x BN(32) x BK(32).
// Waves stack in M (16 rows each); each wave computes a 16x32 strip =
// 2 WMMA tiles sharing one A fragment. A and B are staged into double-buffered
// LDS as bf16 (B stored N-major so a lane's fragment is 16 contiguous bf16 ->
// ds_load_b128). Staging of tile k+1 is issued before computing tile k, so
// global latency overlaps the WMMA pipe; tile k+2 is prefetched
// (global_prefetch_b8). Bounds checks are hoisted to uniform branches.
// ============================================================================
#define BM 128
#define BN 32
#define BK 32
#define LDS_STRIDE (BK + 8)   // pad to 40 elems (80B): keeps b128 16B-aligned, skews banks

__device__ __forceinline__ void stage_tiles(
    __bf16 (*__restrict__ As)[LDS_STRIDE],   // [BM][LDS_STRIDE]
    __bf16 (*__restrict__ Bs)[LDS_STRIDE],   // [BN][LDS_STRIDE]
    const float* __restrict__ A, const float* __restrict__ W,
    int M, int K, int N, int mBase, int nBase, int k0, int tid)
{
    // ---- A tile (BM x BK), coalesced along K ----
    const bool fullA = (mBase + BM <= M) && (k0 + BK <= K) && ((K & 1) == 0);
    if (fullA) {
        // pair-vectorized: float2 load -> packed v2bf store (ds_store_b32)
#pragma unroll
        for (int j = 0; j < (BM * BK / 2) / 256; ++j) {
            const int e  = tid + 256 * j;            // pair index
            const int r  = e >> 4;                   // 16 pairs per row
            const int kp = (e & 15) << 1;
            const float2 v = *reinterpret_cast<const float2*>(
                &A[(size_t)(mBase + r) * K + (k0 + kp)]);
            v2bf pr; pr[0] = (__bf16)v.x; pr[1] = (__bf16)v.y;
            *reinterpret_cast<v2bf*>(&As[r][kp]) = pr;
        }
    } else {
#pragma unroll
        for (int j = 0; j < (BM * BK) / 256; ++j) {
            const int e = tid + 256 * j;
            const int r = e >> 5, k = e & (BK - 1);
            const int rr = mBase + r, kk = k0 + k;
            As[r][k] = (__bf16)((rr < M && kk < K) ? A[(size_t)rr * K + kk] : 0.0f);
        }
    }
    // ---- B tile (BK x BN) -> Bs[n][k], coalesced along N ----
    const bool fullB = (nBase + BN <= N) && (k0 + BK <= K);
    if (fullB) {
#pragma unroll
        for (int j = 0; j < (BN * BK) / 256; ++j) {
            const int e = tid + 256 * j;
            const int k = e >> 5, n = e & (BN - 1);
            Bs[n][k] = (__bf16)W[(size_t)(k0 + k) * N + (nBase + n)];
        }
    } else {
#pragma unroll
        for (int j = 0; j < (BN * BK) / 256; ++j) {
            const int e = tid + 256 * j;
            const int k = e >> 5, n = e & (BN - 1);
            const int kk = k0 + k, nn = nBase + n;
            Bs[n][k] = (__bf16)((kk < K && nn < N) ? W[(size_t)kk * N + nn] : 0.0f);
        }
    }
}

__global__ void gemm_wmma_kernel(const float* __restrict__ A,
                                 const float* __restrict__ W,
                                 const float* __restrict__ bias,
                                 float* __restrict__ C,
                                 int M, int K, int N, int relu)
{
    __shared__ alignas(16) __bf16 Asb[2][BM][LDS_STRIDE];
    __shared__ alignas(16) __bf16 Bsb[2][BN][LDS_STRIDE];

    const int tid   = threadIdx.x;
    const int lane  = tid & 31;
    const int wid   = tid >> 5;              // 0..7, M-strip of this wave
    const int mBase = blockIdx.y * BM;
    const int nBase = blockIdx.x * BN;
    const int half  = lane >> 4;             // 0: lanes 0-15, 1: lanes 16-31
    const int lr    = lane & 15;
    const int rowA  = wid * 16 + lr;         // A-fragment row within block tile

    v8f acc0 = {0.f,0.f,0.f,0.f,0.f,0.f,0.f,0.f};
    v8f acc1 = {0.f,0.f,0.f,0.f,0.f,0.f,0.f,0.f};

    // prime the pipeline with tile 0
    stage_tiles(Asb[0], Bsb[0], A, W, M, K, N, mBase, nBase, 0, tid);
    int cur = 0;

    for (int k0 = 0; k0 < K; k0 += BK) {
        __syncthreads();   // tile 'cur' (for k0) is ready block-wide

        const int kn = k0 + BK;
        if (kn < K) {
            // prefetch tile k0+2*BK toward L2/WGP$ (speculative)
            const int pk = k0 + 2 * BK;
            if (pk < K) {
                const int prow = mBase + (tid >> 1);
                if (prow < M)
                    __builtin_prefetch(&A[(size_t)prow * K + pk + ((tid & 1) << 4)], 0, 0);
            }
            // stage tile k0+BK into the other buffer (overlaps WMMA below)
            stage_tiles(Asb[cur ^ 1], Bsb[cur ^ 1], A, W, M, K, N, mBase, nBase, kn, tid);
        }

        // ---- A fragment (ISA 7.12.2, 16-bit A 16x32): pairs of consecutive K ----
        v16bf a;
#pragma unroll
        for (int j = 0; j < 8; ++j) {
            const int kk = (half << 4) + (j << 1);
            v2bf p = *reinterpret_cast<const v2bf*>(&Asb[cur][rowA][kk]);
            a[2 * j]     = p[0];
            a[2 * j + 1] = p[1];
        }
        // ---- B fragments: lanes hold K = half*16 + e -> 16 contiguous bf16 ----
        const __bf16* bp0 = &Bsb[cur][lr][half * 16];
        const __bf16* bp1 = &Bsb[cur][16 + lr][half * 16];
        v8bf q0 = *reinterpret_cast<const v8bf*>(bp0);
        v8bf q1 = *reinterpret_cast<const v8bf*>(bp0 + 8);
        v8bf q2 = *reinterpret_cast<const v8bf*>(bp1);
        v8bf q3 = *reinterpret_cast<const v8bf*>(bp1 + 8);
        v16bf b0 = __builtin_shufflevector(q0, q1, 0,1,2,3,4,5,6,7,8,9,10,11,12,13,14,15);
        v16bf b1 = __builtin_shufflevector(q2, q3, 0,1,2,3,4,5,6,7,8,9,10,11,12,13,14,15);

        acc0 = __builtin_amdgcn_wmma_f32_16x16x32_bf16(false, a, false, b0,
                                                       (short)0, acc0, false, false);
        acc1 = __builtin_amdgcn_wmma_f32_16x16x32_bf16(false, a, false, b1,
                                                       (short)0, acc1, false, false);
        cur ^= 1;
    }

    // ---- store: D VGPR r -> row = r + half*8, col = lane%16 ----
    const int m0 = mBase + wid * 16;
#pragma unroll
    for (int r = 0; r < 8; ++r) {
        const int rr = m0 + (half << 3) + r;
        if (rr < M) {
            const int c0 = nBase + lr;
            if (c0 < N) {
                float v = acc0[r];
                if (bias) v += bias[c0];
                if (relu) v = fmaxf(v, 0.0f);
                C[(size_t)rr * N + c0] = v;
            }
            const int c1 = nBase + 16 + lr;
            if (c1 < N) {
                float v = acc1[r];
                if (bias) v += bias[c1];
                if (relu) v = fmaxf(v, 0.0f);
                C[(size_t)rr * N + c1] = v;
            }
        }
    }
}

// ============================================================================
// BatchNorm over axis 0 (rows): reduce per channel, then apply.
// ============================================================================
__global__ void bn_reduce_kernel(const float* __restrict__ x, int M, int C,
                                 float* __restrict__ mean, float* __restrict__ var)
{
    const int c = blockIdx.x;
    __shared__ float s1[256], s2[256];
    float a = 0.f, b = 0.f;
    for (int i = threadIdx.x; i < M; i += blockDim.x) {
        float v = x[(size_t)i * C + c];
        a += v; b += v * v;
    }
    s1[threadIdx.x] = a; s2[threadIdx.x] = b;
    __syncthreads();
    for (int s = blockDim.x >> 1; s > 0; s >>= 1) {
        if ((int)threadIdx.x < s) {
            s1[threadIdx.x] += s1[threadIdx.x + s];
            s2[threadIdx.x] += s2[threadIdx.x + s];
        }
        __syncthreads();
    }
    if (threadIdx.x == 0) {
        float m = s1[0] / (float)M;
        mean[c] = m;
        var[c]  = s2[0] / (float)M - m * m;
    }
}

__global__ void bn_apply_kernel(float* __restrict__ x,
                                const float* __restrict__ mean,
                                const float* __restrict__ var,
                                const float* __restrict__ g,
                                const float* __restrict__ b,
                                size_t total, int C, int relu)
{
    size_t i = (size_t)blockIdx.x * blockDim.x + threadIdx.x;
    if (i >= total) return;
    int c = (int)(i % C);
    float y = g[c] * (x[i] - mean[c]) * rsqrtf(var[c] + BN_EPS) + b[c];
    if (relu) y = fmaxf(y, 0.0f);
    x[i] = y;
}

// ============================================================================
// KNN: per-query register top-k over LDS-tiled support set.
// ============================================================================
template<int KK>
__global__ void knn_kernel(const float* __restrict__ sup, int ns,
                           const float* __restrict__ qry, int nq,
                           int* __restrict__ oidx, float* __restrict__ od2)
{
    __shared__ float sx[256], sy[256], sz[256];
    const int q = blockIdx.x * blockDim.x + threadIdx.x;
    const bool act = (q < nq);
    float qx = 0.f, qy = 0.f, qz = 0.f;
    if (act) { qx = qry[3*q]; qy = qry[3*q+1]; qz = qry[3*q+2]; }
    float bd[KK]; int bi[KK];
#pragma unroll
    for (int i = 0; i < KK; ++i) { bd[i] = 3.0e38f; bi[i] = 0; }

    for (int s0 = 0; s0 < ns; s0 += 256) {
        int tile = ns - s0; if (tile > 256) tile = 256;
        __syncthreads();
        for (int t = threadIdx.x; t < tile; t += blockDim.x) {
            sx[t] = sup[3*(s0+t)];
            sy[t] = sup[3*(s0+t)+1];
            sz[t] = sup[3*(s0+t)+2];
        }
        __syncthreads();
        if (act) {
            for (int t = 0; t < tile; ++t) {
                float dx = qx - sx[t], dy = qy - sy[t], dz = qz - sz[t];
                float d = dx*dx + dy*dy + dz*dz;
                if (d < bd[KK-1]) {
                    // insertion: strict '<' keeps earlier index on ties (top_k stable)
#pragma unroll
                    for (int u = KK-1; u >= 0; --u) {
                        bool gt = (u > 0) ? (bd[u-1] > d) : false;
                        if (gt) { bd[u] = bd[u-1]; bi[u] = bi[u-1]; }
                        else    { bd[u] = d; bi[u] = s0 + t; break; }
                    }
                }
            }
        }
    }
    if (act) {
#pragma unroll
        for (int i = 0; i < KK; ++i) {
            oidx[(size_t)q*KK + i] = bi[i];
            od2 [(size_t)q*KK + i] = bd[i];
        }
    }
}

// ============================================================================
// Farthest point sampling: single block, dists resident in LDS, block argmax.
// ============================================================================
__global__ void fps_kernel(const float* __restrict__ xyz, int n, int m,
                           int* __restrict__ out)
{
    __shared__ float dists[8192];
    __shared__ float rv[1024];
    __shared__ int   ri[1024];
    __shared__ int   s_last;
    const int tid = threadIdx.x, nt = blockDim.x;
    for (int i = tid; i < n; i += nt) dists[i] = 1e10f;
    if (tid == 0) { out[0] = 0; s_last = 0; }
    __syncthreads();
    for (int it = 1; it < m; ++it) {
        const int last = s_last;
        const float lx = xyz[3*last], ly = xyz[3*last+1], lz = xyz[3*last+2];
        float best = -1.f; int bidx = 0x7fffffff;
        for (int i = tid; i < n; i += nt) {
            float dx = xyz[3*i] - lx, dy = xyz[3*i+1] - ly, dz = xyz[3*i+2] - lz;
            float d = dx*dx + dy*dy + dz*dz;
            float dd = fminf(dists[i], d);
            dists[i] = dd;
            if (dd > best) { best = dd; bidx = i; }  // strict > keeps first max
        }
        rv[tid] = best; ri[tid] = bidx;
        __syncthreads();
        for (int s = nt >> 1; s > 0; s >>= 1) {
            if (tid < s) {
                if (rv[tid+s] > rv[tid] ||
                    (rv[tid+s] == rv[tid] && ri[tid+s] < ri[tid])) {
                    rv[tid] = rv[tid+s]; ri[tid] = ri[tid+s];
                }
            }
            __syncthreads();
        }
        if (tid == 0) { s_last = ri[0]; out[it] = ri[0]; }
        __syncthreads();
    }
}

// ============================================================================
// Pointwise / gather / reduction helper kernels
// ============================================================================
__global__ void gather_rows_kernel(const float* __restrict__ src,
                                   const int* __restrict__ idx, int idxStride,
                                   float* __restrict__ out, int n, int c)
{
    size_t i = (size_t)blockIdx.x * blockDim.x + threadIdx.x;
    size_t total = (size_t)n * c;
    if (i >= total) return;
    int cc = (int)(i % c); size_t r = i / c;
    out[i] = src[(size_t)idx[r * idxStride] * c + cc];
}

__global__ void rel_kernel(const float* __restrict__ p, const int* __restrict__ idx,
                           float* __restrict__ rel, int n, int ns)
{
    size_t i = (size_t)blockIdx.x * blockDim.x + threadIdx.x;
    size_t total = (size_t)n * ns * 3;
    if (i >= total) return;
    int d = (int)(i % 3); size_t r = i / 3;
    int pt = (int)(r / ns); int nb = idx[r];
    rel[i] = p[3*nb + d] - p[3*pt + d];
}

// r = kf[idx] - q[:,None,:] + pr
__global__ void compute_r_kernel(const float* __restrict__ kf, const int* __restrict__ idx,
                                 const float* __restrict__ q, const float* __restrict__ pr,
                                 float* __restrict__ r, int n, int ns, int c)
{
    size_t i = (size_t)blockIdx.x * blockDim.x + threadIdx.x;
    size_t total = (size_t)n * ns * c;
    if (i >= total) return;
    int cc = (int)(i % c); size_t rr = i / c;
    int pt = (int)(rr / ns); int nb = idx[rr];
    r[i] = kf[(size_t)nb * c + cc] - q[(size_t)pt * c + cc] + pr[i];
}

// softmax over neighbor axis (t in [0,ns)) for each (point, u) pair
__global__ void softmax_ns_kernel(float* __restrict__ w, int n, int ns, int csp)
{
    size_t i = (size_t)blockIdx.x * blockDim.x + threadIdx.x;
    if (i >= (size_t)n * csp) return;
    int u = (int)(i % csp); int pt = (int)(i / csp);
    float mx = -3e38f;
    for (int t = 0; t < ns; ++t)
        mx = fmaxf(mx, w[((size_t)pt*ns + t)*csp + u]);
    float sum = 0.f;
    for (int t = 0; t < ns; ++t)
        sum += expf(w[((size_t)pt*ns + t)*csp + u] - mx);
    float inv = 1.0f / sum;
    for (int t = 0; t < ns; ++t) {
        size_t o = ((size_t)pt*ns + t)*csp + u;
        w[o] = expf(w[o] - mx) * inv;
    }
}

// out[i, s*csp+u] = sum_t (v[idx[i,t], s*csp+u] + pr[i,t,s*csp+u]) * w[i,t,u]
__global__ void attn_agg_kernel(const float* __restrict__ v, const int* __restrict__ idx,
                                const float* __restrict__ pr, const float* __restrict__ w,
                                float* __restrict__ out, int n, int ns, int c, int csp)
{
    size_t i = (size_t)blockIdx.x * blockDim.x + threadIdx.x;
    if (i >= (size_t)n * c) return;
    int cc = (int)(i % c); int pt = (int)(i / c); int u = cc % csp;
    float acc = 0.f;
    for (int t = 0; t < ns; ++t) {
        size_t r = (size_t)pt*ns + t;
        acc += (v[(size_t)idx[r]*c + cc] + pr[r*c + cc]) * w[r*csp + u];
    }
    out[i] = acc;
}

// grouped[i,j,:] = concat(p[idx[i,j]] - np[i], x[idx[i,j]])  (width 3+c)
__global__ void group_td_kernel(const float* __restrict__ p, const float* __restrict__ x,
                                const int* __restrict__ idx, const float* __restrict__ np_,
                                float* __restrict__ out, int m, int ns, int c)
{
    size_t i = (size_t)blockIdx.x * blockDim.x + threadIdx.x;
    const int cw = 3 + c;
    size_t total = (size_t)m * ns * cw;
    if (i >= total) return;
    int d = (int)(i % cw); size_t r = i / cw;
    int pt = (int)(r / ns); int nb = idx[r];
    out[i] = (d < 3) ? (p[3*nb + d] - np_[3*pt + d])
                     : x[(size_t)nb * c + (d - 3)];
}

__global__ void mean_axis1_kernel(const float* __restrict__ y, float* __restrict__ out,
                                  int m, int ns, int c)
{
    size_t i = (size_t)blockIdx.x * blockDim.x + threadIdx.x;
    if (i >= (size_t)m * c) return;
    int cc = (int)(i % c); int pt = (int)(i / c);
    float s = 0.f;
    for (int t = 0; t < ns; ++t) s += y[((size_t)pt*ns + t)*c + cc];
    out[i] = s / (float)ns;
}

__global__ void add_kernel(float* __restrict__ out, const float* __restrict__ a,
                           const float* __restrict__ b, size_t total)
{
    size_t i = (size_t)blockIdx.x * blockDim.x + threadIdx.x;
    if (i < total) out[i] = a[i] + b[i];
}

__global__ void add_relu_inplace_kernel(float* __restrict__ x, const float* __restrict__ z,
                                        size_t total)
{
    size_t i = (size_t)blockIdx.x * blockDim.x + threadIdx.x;
    if (i < total) x[i] = fmaxf(x[i] + z[i], 0.0f);
}

__global__ void concat_bcast_kernel(const float* __restrict__ x, const float* __restrict__ g,
                                    float* __restrict__ out, int n, int c1, int c2)
{
    size_t i = (size_t)blockIdx.x * blockDim.x + threadIdx.x;
    const int cw = c1 + c2;
    if (i >= (size_t)n * cw) return;
    int j = (int)(i % cw); size_t r = i / cw;
    out[i] = (j < c1) ? x[r * c1 + j] : g[j - c1];
}

__global__ void concat4_kernel(const float* __restrict__ a, const float* __restrict__ b,
                               const float* __restrict__ c, const float* __restrict__ d,
                               float* __restrict__ out, int n, int w)
{
    size_t i = (size_t)blockIdx.x * blockDim.x + threadIdx.x;
    const int cw = 4 * w;
    if (i >= (size_t)n * cw) return;
    int j = (int)(i % cw); size_t r = i / cw;
    int sel = j / w, jj = j % w;
    const float* s = (sel == 0) ? a : (sel == 1) ? b : (sel == 2) ? c : d;
    out[i] = s[(size_t)r * w + jj];
}

// 3-NN inverse-distance interpolation
__global__ void interp3_kernel(const float* __restrict__ feat, const int* __restrict__ idx,
                               const float* __restrict__ d2, float* __restrict__ out,
                               int nq, int c)
{
    size_t i = (size_t)blockIdx.x * blockDim.x + threadIdx.x;
    if (i >= (size_t)nq * c) return;
    int cc = (int)(i % c); size_t r = i / c;
    float w0 = 1.0f / (d2[r*3+0] + 1e-8f);
    float w1 = 1.0f / (d2[r*3+1] + 1e-8f);
    float w2 = 1.0f / (d2[r*3+2] + 1e-8f);
    float s = w0 + w1 + w2;
    out[i] = (feat[(size_t)idx[r*3+0]*c + cc] * w0 +
              feat[(size_t)idx[r*3+1]*c + cc] * w1 +
              feat[(size_t)idx[r*3+2]*c + cc] * w2) / s;
}

// ============================================================================
// Host-side orchestration
// ============================================================================
namespace {

struct Scratch { char* base; size_t off; };

static float* salloc(Scratch& s, size_t elems) {
    float* p = (float*)(s.base + s.off);
    s.off += ((elems * sizeof(float)) + 255) & ~(size_t)255;
    return p;
}
static int* salloci(Scratch& s, size_t elems) { return (int*)salloc(s, elems); }

static inline int nblk(size_t total) { return (int)((total + 255) / 256); }

static void gemm(hipStream_t st, const float* A, const float* W, const float* bias,
                 float* C, int M, int K, int N, int relu = 0)
{
    dim3 g((N + BN - 1) / BN, (M + BM - 1) / BM);
    gemm_wmma_kernel<<<g, 256, 0, st>>>(A, W, bias, C, M, K, N, relu);
}

static void bn(hipStream_t st, float* x, const float* g, const float* b,
               int M, int C, int relu, float* stat)
{
    bn_reduce_kernel<<<C, 256, 0, st>>>(x, M, C, stat, stat + 256);
    size_t total = (size_t)M * C;
    bn_apply_kernel<<<nblk(total), 256, 0, st>>>(x, stat, stat + 256, g, b, total, C, relu);
}

static void knn(hipStream_t st, const float* sup, int ns, const float* qry, int nq,
                int k, int* idx, float* d2)
{
    int blocks = (nq + 255) / 256;
    switch (k) {
        case 1:  knn_kernel<1> <<<blocks, 256, 0, st>>>(sup, ns, qry, nq, idx, d2); break;
        case 3:  knn_kernel<3> <<<blocks, 256, 0, st>>>(sup, ns, qry, nq, idx, d2); break;
        case 8:  knn_kernel<8> <<<blocks, 256, 0, st>>>(sup, ns, qry, nq, idx, d2); break;
        default: knn_kernel<16><<<blocks, 256, 0, st>>>(sup, ns, qry, nq, idx, d2); break;
    }
}

struct TrP  { const float *Wq,*bq,*Wk,*bk,*Wv,*bv,*Wp1,*bp1,*gp,*btp,*Wp2,*bp2,
                          *gw1,*btw1,*Ww1,*bw1,*gw2,*btw2,*Ww2,*bw2; };
struct BlkP { const float *W1,*g1,*b1; TrP tr; const float *g2,*b2,*W3,*g3,*b3; };

static void pt_layer(hipStream_t st, Scratch& s, const TrP& P,
                     const float* p, const float* x, float* out,
                     int n, int c, int ns, float* stat)
{
    const int csp = c / 8;
    float* q  = salloc(s, (size_t)n * c);
    float* kf = salloc(s, (size_t)n * c);
    float* v  = salloc(s, (size_t)n * c);
    gemm(st, x, P.Wq, P.bq, q,  n, c, c);
    gemm(st, x, P.Wk, P.bk, kf, n, c, c);
    gemm(st, x, P.Wv, P.bv, v,  n, c, c);

    int*   idx = salloci(s, (size_t)n * ns);
    float* d2  = salloc (s, (size_t)n * ns);
    knn(st, p, n, p, n, ns, idx, d2);

    const size_t R = (size_t)n * ns;
    float* rel = salloc(s, R * 3);
    rel_kernel<<<nblk(R * 3), 256, 0, st>>>(p, idx, rel, n, ns);

    float* pr3 = salloc(s, R * 3);
    gemm(st, rel, P.Wp1, P.bp1, pr3, (int)R, 3, 3);
    bn(st, pr3, P.gp, P.btp, (int)R, 3, 1, stat);

    float* pr = salloc(s, R * c);
    gemm(st, pr3, P.Wp2, P.bp2, pr, (int)R, 3, c);

    float* r = salloc(s, R * c);
    compute_r_kernel<<<nblk(R * c), 256, 0, st>>>(kf, idx, q, pr, r, n, ns, c);
    bn(st, r, P.gw1, P.btw1, (int)R, c, 1, stat);

    float* w1 = salloc(s, R * csp);
    gemm(st, r, P.Ww1, P.bw1, w1, (int)R, c, csp);
    bn(st, w1, P.gw2, P.btw2, (int)R, csp, 1, stat);

    float* w2 = salloc(s, R * csp);
    gemm(st, w1, P.Ww2, P.bw2, w2, (int)R, csp, csp);
    softmax_ns_kernel<<<nblk((size_t)n * csp), 256, 0, st>>>(w2, n, ns, csp);

    attn_agg_kernel<<<nblk((size_t)n * c), 256, 0, st>>>(v, idx, pr, w2, out, n, ns, c, csp);
}

static void bottleneck(hipStream_t st, Scratch& s, const BlkP& P,
                       const float* p, float* x, int n, int c, int ns, float* stat)
{
    size_t mark = s.off;
    float* y = salloc(s, (size_t)n * c);
    gemm(st, x, P.W1, nullptr, y, n, c, c);
    bn(st, y, P.g1, P.b1, n, c, 1, stat);

    float* t = salloc(s, (size_t)n * c);
    pt_layer(st, s, P.tr, p, y, t, n, c, ns, stat);
    bn(st, t, P.g2, P.b2, n, c, 1, stat);

    float* z = salloc(s, (size_t)n * c);
    gemm(st, t, P.W3, nullptr, z, n, c, c);
    bn(st, z, P.g3, P.b3, n, c, 0, stat);

    add_relu_inplace_kernel<<<nblk((size_t)n * c), 256, 0, st>>>(x, z, (size_t)n * c);
    s.off = mark;
}

} // namespace

extern "C" void kernel_launch(void* const* d_in, const int* in_sizes, int n_in,
                              void* d_out, int out_size, void* d_ws, size_t ws_size,
                              hipStream_t stream)
{
    (void)out_size; (void)ws_size;
    const int N1 = 8192, N2 = 2048, N3 = 512;

    // --- locate x0 (unique flat size 8192*3); params consumed sequentially in
    // --- init_params() insertion order, skipping the x0 slot.
    int x0i = 0;
    for (int i = 0; i < n_in; ++i) if (in_sizes[i] == N1 * 3) { x0i = i; break; }
    int cur = 0;
    auto nxt = [&]() -> const float* {
        if (cur == x0i) ++cur;
        return (const float*)d_in[cur++];
    };
    const float* x0 = (const float*)d_in[x0i];

    auto readTr = [&](TrP& t) {
        t.Wq = nxt(); t.bq = nxt(); t.Wk = nxt(); t.bk = nxt(); t.Wv = nxt(); t.bv = nxt();
        t.Wp1 = nxt(); t.bp1 = nxt(); t.gp = nxt(); t.btp = nxt();
        t.Wp2 = nxt(); t.bp2 = nxt();
        t.gw1 = nxt(); t.btw1 = nxt(); t.Ww1 = nxt(); t.bw1 = nxt();
        t.gw2 = nxt(); t.btw2 = nxt(); t.Ww2 = nxt(); t.bw2 = nxt();
    };
    auto readBlk = [&](BlkP& B) {
        B.W1 = nxt(); B.g1 = nxt(); B.b1 = nxt();
        readTr(B.tr);
        B.g2 = nxt(); B.b2 = nxt(); B.W3 = nxt(); B.g3 = nxt(); B.b3 = nxt();
    };

    // enc1
    const float *e1_W = nxt(), *e1_g = nxt(), *e1_b = nxt();
    BlkP blk1; readBlk(blk1);
    // enc2
    const float *e2_W = nxt(), *e2_Wid = nxt(), *e2_g = nxt(), *e2_b = nxt();
    BlkP blk2; readBlk(blk2);
    // enc3
    const float *e3_W = nxt(), *e3_Wid = nxt(), *e3_g = nxt(), *e3_b = nxt();
    BlkP blk3; readBlk(blk3);
    // dec3 (tu_head)
    const float *d3_W1 = nxt(), *d3_b1 = nxt(), *d3_g1 = nxt(), *d3_bt1 = nxt(),
                *d3_W2 = nxt(), *d3_b2 = nxt();
    BlkP dblk3; readBlk(dblk3);
    // dec2 (tu)
    const float *d2_W1 = nxt(), *d2_b1 = nxt(), *d2_g1 = nxt(), *d2_bt1 = nxt(),
                *d2_W2 = nxt(), *d2_b2 = nxt(), *d2_g2 = nxt(), *d2_bt2 = nxt();
    BlkP dblk2; readBlk(dblk2);
    // dec1 (tu)
    const float *d1_W1 = nxt(), *d1_b1 = nxt(), *d1_g1 = nxt(), *d1_bt1 = nxt(),
                *d1_W2 = nxt(), *d1_b2 = nxt(), *d1_g2 = nxt(), *d1_bt2 = nxt();
    BlkP dblk1; readBlk(dblk1);
    // up x3
    const float *up_W1[3], *up_b1[3], *up_g[3], *up_bt[3], *up_W2[3], *up_b2[3];
    for (int l = 0; l < 3; ++l) {
        up_W1[l] = nxt(); up_b1[l] = nxt(); up_g[l] = nxt(); up_bt[l] = nxt();
        up_W2[l] = nxt(); up_b2[l] = nxt();
    }
    // cls
    const float *c_W1 = nxt(), *c_b1 = nxt(), *c_g = nxt(), *c_bt = nxt(),
                *c_W2 = nxt(), *c_b2 = nxt();

    // --- persistent workspace ---
    Scratch s{ (char*)d_ws, 0 };
    float* stat = salloc(s, 512);                 // bn mean/var scratch (reused)
    float* x1 = salloc(s, (size_t)N1 * 32);
    float* p2 = salloc(s, (size_t)N2 * 3);
    float* x2 = salloc(s, (size_t)N2 * 64);
    float* p3 = salloc(s, (size_t)N3 * 3);
    float* x3 = salloc(s, (size_t)N3 * 128);
    float* f1 = salloc(s, (size_t)N1 * 32);
    float* f2 = salloc(s, (size_t)N1 * 32);
    float* f3 = salloc(s, (size_t)N1 * 32);
    float* d3 = salloc(s, (size_t)N3 * 128);
    float* d2f = salloc(s, (size_t)N2 * 64);
    float* d1f = salloc(s, (size_t)N1 * 32);
    const size_t baseMark = s.off;
    hipStream_t st = stream;

    // ================= enc1 (stride 1) =================
    gemm(st, x0, e1_W, nullptr, x1, N1, 3, 32);
    bn(st, x1, e1_g, e1_b, N1, 32, 1, stat);
    bottleneck(st, s, blk1, /*p1=*/x0, x1, N1, 32, 8, stat);

    // ================= enc2 (stride 4, ns 16) =================
    {
        s.off = baseMark;
        int* fidx = salloci(s, N2);
        fps_kernel<<<1, 1024, 0, st>>>(x0, N1, N2, fidx);
        gather_rows_kernel<<<nblk((size_t)N2*3), 256, 0, st>>>(x0, fidx, 1, p2, N2, 3);
        float* ident = salloc(s, (size_t)N2 * 32);
        gather_rows_kernel<<<nblk((size_t)N2*32), 256, 0, st>>>(x1, fidx, 1, ident, N2, 32);
        int*   kidx = salloci(s, (size_t)N2 * 16);
        float* kd2  = salloc (s, (size_t)N2 * 16);
        knn(st, x0, N1, p2, N2, 16, kidx, kd2);
        float* grouped = salloc(s, (size_t)N2 * 16 * 35);
        group_td_kernel<<<nblk((size_t)N2*16*35), 256, 0, st>>>(x0, x1, kidx, p2, grouped, N2, 16, 32);
        float* y = salloc(s, (size_t)N2 * 16 * 64);
        gemm(st, grouped, e2_W, nullptr, y, N2*16, 35, 64);
        bn(st, y, e2_g, e2_b, N2*16, 64, 1, stat);
        float* ym = salloc(s, (size_t)N2 * 64);
        mean_axis1_kernel<<<nblk((size_t)N2*64), 256, 0, st>>>(y, ym, N2, 16, 64);
        float* y2 = salloc(s, (size_t)N2 * 64);
        gemm(st, ident, e2_Wid, nullptr, y2, N2, 32, 64);
        bn(st, y2, e2_g, e2_b, N2, 64, 1, stat);
        add_kernel<<<nblk((size_t)N2*64), 256, 0, st>>>(x2, ym, y2, (size_t)N2*64);
        bottleneck(st, s, blk2, p2, x2, N2, 64, 16, stat);
    }

    // ================= enc3 (stride 4, ns 16) =================
    {
        s.off = baseMark;
        int* fidx = salloci(s, N3);
        fps_kernel<<<1, 1024, 0, st>>>(p2, N2, N3, fidx);
        gather_rows_kernel<<<nblk((size_t)N3*3), 256, 0, st>>>(p2, fidx, 1, p3, N3, 3);
        float* ident = salloc(s, (size_t)N3 * 64);
        gather_rows_kernel<<<nblk((size_t)N3*64), 256, 0, st>>>(x2, fidx, 1, ident, N3, 64);
        int*   kidx = salloci(s, (size_t)N3 * 16);
        float* kd2  = salloc (s, (size_t)N3 * 16);
        knn(st, p2, N2, p3, N3, 16, kidx, kd2);
        float* grouped = salloc(s, (size_t)N3 * 16 * 67);
        group_td_kernel<<<nblk((size_t)N3*16*67), 256, 0, st>>>(p2, x2, kidx, p3, grouped, N3, 16, 64);
        float* y = salloc(s, (size_t)N3 * 16 * 128);
        gemm(st, grouped, e3_W, nullptr, y, N3*16, 67, 128);
        bn(st, y, e3_g, e3_b, N3*16, 128, 1, stat);
        float* ym = salloc(s, (size_t)N3 * 128);
        mean_axis1_kernel<<<nblk((size_t)N3*128), 256, 0, st>>>(y, ym, N3, 16, 128);
        float* y2 = salloc(s, (size_t)N3 * 128);
        gemm(st, ident, e3_Wid, nullptr, y2, N3, 64, 128);
        bn(st, y2, e3_g, e3_b, N3, 128, 1, stat);
        add_kernel<<<nblk((size_t)N3*128), 256, 0, st>>>(x3, ym, y2, (size_t)N3*128);
        bottleneck(st, s, blk3, p3, x3, N3, 128, 16, stat);
    }

    // ================= up-branches (1-NN interp to p0, MLP) =================
    {
        const float* pe[3] = { x0, p2, p3 };
        const float* xe[3] = { x1, x2, x3 };
        const int    ne[3] = { N1, N2, N3 };
        const int    ce[3] = { 32, 64, 128 };
        float* feats[3] = { f1, f2, f3 };
        for (int l = 0; l < 3; ++l) {
            s.off = baseMark;
            int*   i1 = salloci(s, N1);
            float* dd = salloc (s, N1);
            knn(st, pe[l], ne[l], x0, N1, 1, i1, dd);
            float* f = salloc(s, (size_t)N1 * ce[l]);
            gather_rows_kernel<<<nblk((size_t)N1*ce[l]), 256, 0, st>>>(xe[l], i1, 1, f, N1, ce[l]);
            float* ft = salloc(s, (size_t)N1 * ce[l]);
            gemm(st, f, up_W1[l], up_b1[l], ft, N1, ce[l], ce[l]);
            bn(st, ft, up_g[l], up_bt[l], N1, ce[l], 1, stat);
            gemm(st, ft, up_W2[l], up_b2[l], feats[l], N1, ce[l], 32);
        }
    }

    // ================= dec3 (tu_head + bottleneck) =================
    {
        s.off = baseMark;
        bn_reduce_kernel<<<128, 256, 0, st>>>(x3, N3, 128, stat, stat + 256); // mean(x3,0) -> stat
        float* gvec = salloc(s, 128);
        gemm(st, stat, d3_W2, d3_b2, gvec, 1, 128, 128, /*relu=*/1);
        float* ycat = salloc(s, (size_t)N3 * 256);
        concat_bcast_kernel<<<nblk((size_t)N3*256), 256, 0, st>>>(x3, gvec, ycat, N3, 128, 128);
        gemm(st, ycat, d3_W1, d3_b1, d3, N3, 256, 128);
        bn(st, d3, d3_g1, d3_bt1, N3, 128, 1, stat);
        bottleneck(st, s, dblk3, p3, d3, N3, 128, 16, stat);
    }

    // ================= dec2 (tu + bottleneck) =================
    {
        s.off = baseMark;
        float* a = salloc(s, (size_t)N2 * 64);
        gemm(st, x2, d2_W1, d2_b1, a, N2, 64, 64);
        bn(st, a, d2_g1, d2_bt1, N2, 64, 1, stat);
        float* bb = salloc(s, (size_t)N3 * 64);
        gemm(st, d3, d2_W2, d2_b2, bb, N3, 128, 64);
        bn(st, bb, d2_g2, d2_bt2, N3, 64, 1, stat);
        int*   i3 = salloci(s, (size_t)N2 * 3);
        float* dd = salloc (s, (size_t)N2 * 3);
        knn(st, p3, N3, p2, N2, 3, i3, dd);
        float* bi = salloc(s, (size_t)N2 * 64);
        interp3_kernel<<<nblk((size_t)N2*64), 256, 0, st>>>(bb, i3, dd, bi, N2, 64);
        add_kernel<<<nblk((size_t)N2*64), 256, 0, st>>>(d2f, a, bi, (size_t)N2*64);
        bottleneck(st, s, dblk2, p2, d2f, N2, 64, 16, stat);
    }

    // ================= dec1 (tu + bottleneck) =================
    {
        s.off = baseMark;
        float* a = salloc(s, (size_t)N1 * 32);
        gemm(st, x1, d1_W1, d1_b1, a, N1, 32, 32);
        bn(st, a, d1_g1, d1_bt1, N1, 32, 1, stat);
        float* bb = salloc(s, (size_t)N2 * 32);
        gemm(st, d2f, d1_W2, d1_b2, bb, N2, 64, 32);
        bn(st, bb, d1_g2, d1_bt2, N2, 32, 1, stat);
        int*   i3 = salloci(s, (size_t)N1 * 3);
        float* dd = salloc (s, (size_t)N1 * 3);
        knn(st, p2, N2, x0, N1, 3, i3, dd);
        float* bi = salloc(s, (size_t)N1 * 32);
        interp3_kernel<<<nblk((size_t)N1*32), 256, 0, st>>>(bb, i3, dd, bi, N1, 32);
        add_kernel<<<nblk((size_t)N1*32), 256, 0, st>>>(d1f, a, bi, (size_t)N1*32);
        bottleneck(st, s, dblk1, x0, d1f, N1, 32, 8, stat);
    }

    // ================= classifier head =================
    {
        s.off = baseMark;
        float* h = salloc(s, (size_t)N1 * 128);
        concat4_kernel<<<nblk((size_t)N1*128), 256, 0, st>>>(f1, f2, f3, d1f, h, N1, 32);
        float* h2 = salloc(s, (size_t)N1 * 32);
        gemm(st, h, c_W1, c_b1, h2, N1, 128, 32);
        bn(st, h2, c_g, c_bt, N1, 32, 1, stat);
        gemm(st, h2, c_W2, c_b2, (float*)d_out, N1, 32, 2);
    }
}
